// FocalLoss_74758200754691
// MI455X (gfx1250) — compile-verified, matrix operand order checked by reading, not run on hardware
//
#include <hip/hip_runtime.h>
#include <hip/hip_bf16.h>

typedef __attribute__((ext_vector_type(16))) _Float16 v16h;
typedef __attribute__((ext_vector_type(8)))  float    v8f;

#define WAVE 32
#define NEG_FLT_MAX (-3.402823466e+38f)

// ---------------------------------------------------------------------------
// Kernel 1: one wave32 per row. Single pass over the 1 GiB logits (HBM-bound).
// Each lane owns 32 contiguous-by-vector floats (8x float4, coalesced B128).
// Online softmax merge across lanes via shfl_xor. Writes (loss, valid) per row.
// ---------------------------------------------------------------------------
__global__ __launch_bounds__(256) void focal_rows_kernel(
    const float* __restrict__ x, const int* __restrict__ tgt,
    float2* __restrict__ rowout, int N, int C /* == 1024 */) {
  const int lane = threadIdx.x & (WAVE - 1);
  const int wavei = threadIdx.x >> 5;
  const int row = blockIdx.x * 8 + wavei;
  if (row >= N) return;  // wave-uniform guard

  const float* __restrict__ xr = x + (size_t)row * (size_t)C;
  const float4* __restrict__ x4 = reinterpret_cast<const float4*>(xr);

  // Target is wave-uniform -> scalar load; prefetch the gathered logit early
  // so the dependent 4B load at the end hits cache (gfx1250 global_prefetch_b8).
  const int t = tgt[row];
  if (t >= 0) __builtin_prefetch(xr + t, 0, 3);

  // Load all 32 values for this lane into registers (8 x global_load_b128).
  float4 v[8];
#pragma unroll
  for (int j = 0; j < 8; ++j) v[j] = x4[j * WAVE + lane];

  // Lane-local max.
  float m = NEG_FLT_MAX;
#pragma unroll
  for (int j = 0; j < 8; ++j) {
    m = fmaxf(m, fmaxf(fmaxf(v[j].x, v[j].y), fmaxf(v[j].z, v[j].w)));
  }
  // Lane-local sum of exp(v - m).
  float s = 0.0f;
#pragma unroll
  for (int j = 0; j < 8; ++j) {
    s += __expf(v[j].x - m) + __expf(v[j].y - m) +
         __expf(v[j].z - m) + __expf(v[j].w - m);
  }

  // Cross-lane merge of (m, s) pairs with rescaling: 5 steps on wave32.
#pragma unroll
  for (int off = 16; off > 0; off >>= 1) {
    float om = __shfl_xor(m, off, WAVE);
    float os = __shfl_xor(s, off, WAVE);
    float nm = fmaxf(m, om);
    s = s * __expf(m - nm) + os * __expf(om - nm);
    m = nm;
  }

  if (lane == 0) {
    float2 r;
    if (t >= 0) {
      const float xt    = xr[t];
      const float lse   = m + logf(s);
      const float logpt = xt - lse;
      const float pt    = __expf(logpt);
      const float omp   = 1.0f - pt;
      r.x = -(omp * omp) * logpt;  // GAMMA = 2.0
      r.y = 1.0f;
    } else {
      r.x = 0.0f;  // IGNORE_INDEX
      r.y = 0.0f;
    }
    rowout[row] = r;
  }
}

// ---------------------------------------------------------------------------
// Kernel 2: WMMA tree reduction. Each wave reduces 512 (loss, valid) pairs
// using v_wmma_f32_16x16x32_f16 with B == ones. For any lane->(m,k) bijection,
// D[m][n] = rowsum_m(A), hence sum of all D entries == 16 * sum(A): the result
// is exact independent of the ISA's A/B VGPR striping. 8 waves/block -> one
// float2 partial per block (fixed order => deterministic).
// ---------------------------------------------------------------------------
__global__ __launch_bounds__(256) void focal_reduce2_kernel(
    const float2* __restrict__ in, float2* __restrict__ part) {
  const int lane = threadIdx.x & (WAVE - 1);
  const int wavei = threadIdx.x >> 5;
  const size_t base = ((size_t)blockIdx.x * 8 + wavei) * 512;

  v16h aL, aC, ones;
#pragma unroll
  for (int k = 0; k < 16; ++k) {
    float2 p = in[base + (size_t)k * WAVE + lane];
    aL[k] = (_Float16)p.x;
    aC[k] = (_Float16)p.y;
    ones[k] = (_Float16)1.0f;
  }

  v8f zero = {0.0f, 0.0f, 0.0f, 0.0f, 0.0f, 0.0f, 0.0f, 0.0f};
  // EXEC is all-ones here (no divergence before this point in the block).
  v8f dL = __builtin_amdgcn_wmma_f32_16x16x32_f16(
      false, aL, false, ones, (short)0, zero, false, false);
  v8f dC = __builtin_amdgcn_wmma_f32_16x16x32_f16(
      false, aC, false, ones, (short)0, zero, false, false);

  float tl = 0.0f, tc = 0.0f;
#pragma unroll
  for (int r = 0; r < 8; ++r) { tl += dL[r]; tc += dC[r]; }
#pragma unroll
  for (int off = 16; off > 0; off >>= 1) {
    tl += __shfl_xor(tl, off, WAVE);
    tc += __shfl_xor(tc, off, WAVE);
  }
  tl *= (1.0f / 16.0f);  // ones-trick replicates each rowsum across 16 columns
  tc *= (1.0f / 16.0f);

  __shared__ float2 sm[8];
  if (lane == 0) { sm[wavei].x = tl; sm[wavei].y = tc; }
  __syncthreads();
  if (threadIdx.x == 0) {
    float L = 0.0f, Cn = 0.0f;
#pragma unroll
    for (int w = 0; w < 8; ++w) { L += sm[w].x; Cn += sm[w].y; }
    float2 r; r.x = L; r.y = Cn;
    part[blockIdx.x] = r;
  }
}

// ---------------------------------------------------------------------------
// Kernel 3: final 64 partials -> scalar mean. Single tiny block.
// ---------------------------------------------------------------------------
__global__ void focal_final_kernel(const float2* __restrict__ part,
                                   float* __restrict__ out, int n) {
  const int lane = threadIdx.x & (WAVE - 1);
  const int wavei = threadIdx.x >> 5;
  float L = 0.0f, Cn = 0.0f;
  if ((int)threadIdx.x < n) { float2 p = part[threadIdx.x]; L = p.x; Cn = p.y; }
#pragma unroll
  for (int off = 16; off > 0; off >>= 1) {
    L += __shfl_xor(L, off, WAVE);
    Cn += __shfl_xor(Cn, off, WAVE);
  }
  __shared__ float2 sm[2];
  if (lane == 0) { sm[wavei].x = L; sm[wavei].y = Cn; }
  __syncthreads();
  if (threadIdx.x == 0) {
    float tL = sm[0].x + sm[1].x;
    float tC = sm[0].y + sm[1].y;
    out[0] = tL / fmaxf(tC, 1.0f);  // SIZE_AVERAGE
  }
}

extern "C" void kernel_launch(void* const* d_in, const int* in_sizes, int n_in,
                              void* d_out, int out_size, void* d_ws, size_t ws_size,
                              hipStream_t stream) {
  const float* x  = (const float*)d_in[0];
  const int* tgt  = (const int*)d_in[1];
  float* out      = (float*)d_out;

  const int N = in_sizes[1];            // 262144
  const int C = in_sizes[0] / N;        // 1024

  float2* rowbuf = (float2*)d_ws;                                   // N float2
  float2* part   = (float2*)((char*)d_ws + (size_t)N * sizeof(float2));

  const int blocks1 = (N + 7) / 8;        // 8 waves (rows) per 256-thread block
  const int blocks2 = N / 4096;           // 8 waves x 512 pairs per block = 64
  focal_rows_kernel<<<blocks1, 256, 0, stream>>>(x, tgt, rowbuf, N, C);
  focal_reduce2_kernel<<<blocks2, 256, 0, stream>>>(rowbuf, part);
  focal_final_kernel<<<1, 64, 0, stream>>>(part, out, blocks2);
}
